// LoRALayer_11836929868064
// MI455X (gfx1250) — matrix operand order repaired and checked by reading, not compiled, not run on hardware
//
#include <hip/hip_runtime.h>
#include <cstdint>

// ---------------------------------------------------------------------------
// LoRA forward on MI455X (gfx1250), folded into a single bf16 WMMA GEMM:
//   W_eff = W + 2.0 * lora_B @ lora_A          (rank-16 fold, ~0.5 GFLOP)
//   out   = x @ W_eff.T + b                    (275 GFLOP, tensor cores)
// M=8192, N=4096, K=4096.  Compute-bound -> bf16 A/B, fp32 accumulate.
// Data path: TDM tensor_load_to_lds double-buffering (TENSORcnt), WMMA from LDS.
// ---------------------------------------------------------------------------

typedef __attribute__((ext_vector_type(16))) __bf16 v16bf;
typedef __attribute__((ext_vector_type(8)))  float  v8f;
typedef __attribute__((ext_vector_type(4)))  unsigned int v4u;
typedef __attribute__((ext_vector_type(8)))  int    v8i;
typedef __attribute__((ext_vector_type(4)))  int    v4i;

struct __align__(16) U4 { unsigned int x, y, z, w; };
struct __align__(8)  US4 { unsigned short x, y, z, w; };

union Frag32 {              // 32 bytes: two 16B chunks <-> one v16bf WMMA operand
    U4    q[2];
    v16bf v;
};

__device__ __forceinline__ unsigned short f2bf(float f) {
    unsigned int u = __float_as_uint(f);
    u += 0x7FFFu + ((u >> 16) & 1u);      // round-to-nearest-even
    return (unsigned short)(u >> 16);
}

// ---------------- stage 1: x (fp32) -> bf16 ---------------------------------
__global__ __launch_bounds__(256) void k_convert_x(const float* __restrict__ x,
                                                   unsigned short* __restrict__ xb,
                                                   int n4) {
    int i = blockIdx.x * 256 + threadIdx.x;
    if (i >= n4) return;
    float4 v = reinterpret_cast<const float4*>(x)[i];
    US4 o;
    o.x = f2bf(v.x); o.y = f2bf(v.y); o.z = f2bf(v.z); o.w = f2bf(v.w);
    reinterpret_cast<US4*>(xb)[i] = o;
}

// ---------------- stage 2: W_eff = W + 2*(B@A), fp32 -> bf16 ----------------
__global__ __launch_bounds__(256) void k_fuse_w(const float* __restrict__ W,
                                                const float* __restrict__ lA,   // [16,4096]
                                                const float* __restrict__ lB,   // [4096,16]
                                                unsigned short* __restrict__ wb) {
    int idx = blockIdx.x * 256 + threadIdx.x;     // one thread per 4 outputs
    int o   = idx >> 10;                          // 4096/4 = 1024 quads per row
    int i4  = (idx & 1023);
    float bv[16];
#pragma unroll
    for (int r = 0; r < 16; ++r) bv[r] = lB[o * 16 + r];
    const float4* lA4 = reinterpret_cast<const float4*>(lA);
    float4 w = reinterpret_cast<const float4*>(W)[o * 1024 + i4];
    float4 s = make_float4(0.f, 0.f, 0.f, 0.f);
#pragma unroll
    for (int r = 0; r < 16; ++r) {
        float4 a = lA4[r * 1024 + i4];
        s.x += bv[r] * a.x; s.y += bv[r] * a.y;
        s.z += bv[r] * a.z; s.w += bv[r] * a.w;
    }
    const float SCALING = 2.0f;                   // ALPHA/RANK = 32/16
    US4 out;
    out.x = f2bf(w.x + SCALING * s.x);
    out.y = f2bf(w.y + SCALING * s.y);
    out.z = f2bf(w.z + SCALING * s.z);
    out.w = f2bf(w.w + SCALING * s.w);
    reinterpret_cast<US4*>(wb)[o * 1024 + i4] = out;
}

// ---------------- TDM descriptor build + issue ------------------------------
// 2-D bf16 tile (tile_d0 x tile_d1 elements) from row-major global (stride
// elements) into LDS at lds_byte, with HW padding: every 16 DWORDs (64 B = one
// 32-elem row) insert 4 DWORDs (16 B) -> LDS row stride 80 B (bank-spread).
__device__ __forceinline__ void tdm_load_tile(unsigned lds_byte,
                                              const unsigned short* gptr,
                                              unsigned tile_d0, unsigned tile_d1,
                                              unsigned long stride_elems) {
    unsigned long ga = (unsigned long)(uintptr_t)gptr;
    v4u g0;
    g0[0] = 1u;                                          // count=1 valid user D#
    g0[1] = lds_byte;                                    // lds_addr
    g0[2] = (unsigned)(ga & 0xFFFFFFFFu);                // global_addr[31:0]
    g0[3] = (unsigned)((ga >> 32) & 0x01FFFFFFu)         // global_addr[56:32]
          | 0x80000000u;                                 // type=2 ("image")
    v8i g1;
    g1[0] = (int)((1u << 16)      // data_size = 1 -> 2 bytes
                | (1u << 20)      // pad_enable
                | (3u << 22)      // pad_interval: 16 DWORDs
                | (3u << 25));    // pad_amount:   4 DWORDs
    g1[1] = (int)((tile_d0 & 0xFFFFu) << 16);            // tensor_dim0 = tile_d0
    g1[2] = (int)((tile_d0 >> 16) | ((tile_d1 & 0xFFFFu) << 16)); // tensor_dim1
    g1[3] = (int)((tile_d1 >> 16) | (tile_d0 << 16));    // tile_dim0
    g1[4] = (int)(tile_d1);                              // tile_dim1; tile_dim2=0
    g1[5] = (int)(stride_elems & 0xFFFFFFFFu);           // tensor_dim0_stride lo
    g1[6] = (int)((stride_elems >> 32) & 0xFFFFu);       // stride hi | dim1_stride lo
    g1[7] = 0;
    v4i z4{};
    v8i z8{};
    // 6-arg form (clang-23 / therock-10.0 headers): groups 0..4 + cache policy
    __builtin_amdgcn_tensor_load_to_lds(g0, g1, z4, z4, z8, 0);
}

// ---------------- stage 3: out = Xb @ Wb^T + bias (bf16 WMMA) ---------------
// Tile: 128x128 per workgroup, BK=32. 256 threads = 8 waves (wave32),
// wave grid 2(M) x 4(N); each wave computes 64x32 = 4x2 tiles of 16x16.
// TDM double-buffers the 128x32 A/B tiles in LDS (row stride 5*U4 = 80 B),
// buffers rotated by pointer swap so the loop body is register-stable.
__global__ __launch_bounds__(256) void k_lora_gemm(const unsigned short* __restrict__ Xb,
                                                   const unsigned short* __restrict__ Wb,
                                                   const float* __restrict__ bias,
                                                   float* __restrict__ out) {
    constexpr int Kd = 4096, Nd = 4096;
    constexpr int KT = Kd / 32;                  // 128 K-steps

    __shared__ U4 As[2][128][5];                 // 80 B row stride (64 data + 16 pad)
    __shared__ U4 Bs[2][128][5];                 // total 40 KB of the 320 KB WGP pool

    const int tid   = threadIdx.x;
    const int lane  = tid & 31;
    const int wave  = tid >> 5;
    const int waveM = wave >> 2;                 // 0..1 -> M offset 0/64
    const int waveN = wave & 3;                  // 0..3 -> N offset 0/32/64/96
    const int lrow  = lane & 15;
    const bool hi   = lane >= 16;
    const bool w0   = (wave == 0);

    const int mBase = blockIdx.y * 128;
    const int nBase = blockIdx.x * 128;

    // double-buffer state, rotated by swap (no data-dependent indexing)
    const U4 (*rdA)[5] = As[0];
    const U4 (*rdB)[5] = Bs[0];
    unsigned rdLdsA = (unsigned)(uintptr_t)&As[0][0][0];
    unsigned rdLdsB = (unsigned)(uintptr_t)&Bs[0][0][0];
    const U4 (*wrA)[5] = As[1];
    const U4 (*wrB)[5] = Bs[1];
    unsigned wrLdsA = (unsigned)(uintptr_t)&As[1][0][0];
    unsigned wrLdsB = (unsigned)(uintptr_t)&Bs[1][0][0];

    const unsigned short* gA = Xb + (size_t)mBase * Kd;   // advance by 32 per step
    const unsigned short* gB = Wb + (size_t)nBase * Kd;

    v8f acc[4][2];
#pragma unroll
    for (int i = 0; i < 4; ++i)
#pragma unroll
        for (int j = 0; j < 2; ++j) acc[i][j] = v8f{};

    // A frag (16-bit 16x32): lane<16 -> K{0..7,16..23} = U4 idx {0,2};
    //                        lane>=16 -> K{8..15,24..31} = U4 idx {1,3}.
    const int a0 = hi ? 1 : 0, a1 = hi ? 3 : 2;
    // B frag (16-bit 32x16): lane<16 -> K{0..7,16..23} (U4 0,2); lane>=16 -> K{8..15,24..31} (U4 1,3).
    const int b0 = hi ? 1 : 0;

    if (w0) {                                    // preload tile 0 into read buffer
        tdm_load_tile(rdLdsA, gA, 32u, 128u, (unsigned long)Kd);
        tdm_load_tile(rdLdsB, gB, 32u, 128u, (unsigned long)Kd);
    }
    __builtin_amdgcn_s_wait_tensorcnt(0);        // no-op for waves 1..7
    __syncthreads();

    for (int kt = 0; kt < KT; ++kt) {
        if (w0 && (kt + 1) < KT) {               // DMA next tile into write buffer
            tdm_load_tile(wrLdsA, gA + (size_t)(kt + 1) * 32, 32u, 128u, (unsigned long)Kd);
            tdm_load_tile(wrLdsB, gB + (size_t)(kt + 1) * 32, 32u, 128u, (unsigned long)Kd);
        }

        // B fragments once (16 VGPRs), stream A fragments per M-subtile.
        const U4* pb0 = rdB[waveN * 32 + lrow];
        const U4* pb1 = rdB[waveN * 32 + 16 + lrow];
        Frag32 fb0, fb1;
        fb0.q[0] = pb0[b0]; fb0.q[1] = pb0[b0 + 2];
        fb1.q[0] = pb1[b0]; fb1.q[1] = pb1[b0 + 2];
#pragma unroll
        for (int i = 0; i < 4; ++i) {
            const U4* pa = rdA[waveM * 64 + i * 16 + lrow];
            Frag32 fa;
            fa.q[0] = pa[a0];
            fa.q[1] = pa[a1];
            acc[i][0] = __builtin_amdgcn_wmma_f32_16x16x32_bf16(
                false, fa.v, false, fb0.v, (short)0, acc[i][0], false, false);
            acc[i][1] = __builtin_amdgcn_wmma_f32_16x16x32_bf16(
                false, fa.v, false, fb1.v, (short)0, acc[i][1], false, false);
        }

        __builtin_amdgcn_s_wait_tensorcnt(0);    // DMA for kt+1 done (no-op waves 1..7)
        __syncthreads();                         // publish buffer swap

        { const U4 (*t)[5] = rdA; rdA = wrA; wrA = t; }
        { const U4 (*t)[5] = rdB; rdB = wrB; wrB = t; }
        { unsigned t = rdLdsA; rdLdsA = wrLdsA; wrLdsA = t; }
        { unsigned t = rdLdsB; rdLdsB = wrLdsB; wrLdsB = t; }
    }

    // Epilogue: C/D layout -> lanes 0-15: M=v, lanes 16-31: M=v+8; N=lane&15.
#pragma unroll
    for (int i = 0; i < 4; ++i) {
        const int mrow = mBase + waveM * 64 + i * 16 + (hi ? 8 : 0);
#pragma unroll
        for (int j = 0; j < 2; ++j) {
            const int col = nBase + waveN * 32 + j * 16 + lrow;
            const float bn = bias[col];
#pragma unroll
            for (int v = 0; v < 8; ++v)
                out[(size_t)(mrow + v) * Nd + col] = acc[i][j][v] + bn;
        }
    }
}

// ---------------------------------------------------------------------------
extern "C" void kernel_launch(void* const* d_in, const int* in_sizes, int n_in,
                              void* d_out, int out_size, void* d_ws, size_t ws_size,
                              hipStream_t stream) {
    const float* x  = (const float*)d_in[0];   // [4,2048,4096]
    const float* W  = (const float*)d_in[1];   // [4096,4096]
    const float* b  = (const float*)d_in[2];   // [4096]
    const float* lA = (const float*)d_in[3];   // [16,4096]
    const float* lB = (const float*)d_in[4];   // [4096,16]
    float* out = (float*)d_out;                // [4,2048,4096]

    const size_t M = 8192, N = 4096, K = 4096;

    unsigned short* Xb = (unsigned short*)d_ws;                       // 64 MiB
    unsigned short* Wb = (unsigned short*)((char*)d_ws + M * K * 2);  // +32 MiB

    // x fp32 -> bf16
    {
        int n4 = (int)(M * K / 4);
        k_convert_x<<<n4 / 256, 256, 0, stream>>>(x, Xb, n4);
    }
    // W_eff = W + 2*(B@A) fused fold, fp32 -> bf16
    {
        int threads = (int)(N * K / 4);
        k_fuse_w<<<threads / 256, 256, 0, stream>>>(W, lA, lB, Wb);
    }
    // single bf16 WMMA GEMM + bias, TDM-fed
    {
        dim3 grid((unsigned)(N / 128), (unsigned)(M / 128));
        k_lora_gemm<<<grid, 256, 0, stream>>>(Xb, Wb, b, out);
    }
}
